// ST_Transformer_adaptive_super00_515396075932
// MI455X (gfx1250) — matrix operand order repaired and checked by reading, not compiled, not run on hardware
//
#include <hip/hip_runtime.h>
#include <math.h>

typedef __attribute__((ext_vector_type(16))) __bf16 bf16x16;
typedef __attribute__((ext_vector_type(8)))  __bf16 bf16x8;
typedef __attribute__((ext_vector_type(4)))  __bf16 bf16x4;
typedef __attribute__((ext_vector_type(8)))  float  f32x8;

// ---------------- problem constants ----------------
constexpr int   Nn   = 6000;
constexpr int   Ee   = 96000;
constexpr int   DINc = 3000;
constexpr int   FH1c = 256;
constexpr int   FH2c = 128;
constexpr int   GHc  = 128;
constexpr int   LATc = 256;
constexpr int   KT   = 10;
constexpr int   SCLc = 9;
constexpr int   NCc  = 10;
constexpr float ATc  = 0.5f;
constexpr float BNE  = 1e-4f;

constexpr int TM = 64, TN = 64, TK = 32;

static inline int cdiv(long a, long b) { return (int)((a + b - 1) / b); }

// ---------------- helpers ----------------
__device__ __forceinline__ unsigned f2ord(float f) {
  unsigned u = __float_as_uint(f);
  return (u & 0x80000000u) ? ~u : (u | 0x80000000u);
}
__device__ __forceinline__ float ord2f(unsigned u) {
  unsigned v = (u & 0x80000000u) ? (u & 0x7FFFFFFFu) : ~u;
  return __uint_as_float(v);
}

// ---------------- WMMA GEMM ----------------
// A: f32 [M,K] row-major (converted to bf16 while staging).
// B: bf16 [Nc,K] row-major (pre-converted / pre-transposed), i.e. C = A @ B^T.
// MODE 0: C = A@Bt + bias
// MODE 1: C = elu(bn(A@Bt + bias, gamma, beta))   (optionally duplicated to C2)
// MODE 2: C = 2*(A@Bt) - rowN[i] - colN[j] - |sp_i - sp_j|^2   (== -(df+ds))
//
// 64x64 tile, 4 waves (each 32x32 = 2x2 WMMA subtiles), K-step 32.
// Register double-buffered staging: next tile's global loads are issued before
// the WMMAs of the current tile; stores go to the alternate LDS buffer.
template<int MODE>
__global__ __launch_bounds__(128)
void gemm_wmma(const float* __restrict__ A, const __bf16* __restrict__ B,
               const float* __restrict__ bias, const float* __restrict__ gamma,
               const float* __restrict__ beta,
               float* __restrict__ C, float* __restrict__ C2,
               const float* __restrict__ rowN, const float* __restrict__ colN,
               const float* __restrict__ sp,
               int M, int Kd, int Nc)
{
  __shared__ __bf16 Al[2][TM][TK];     // A tile [m][k]
  __shared__ __bf16 Bl[2][TN][TK];     // B tile [n][k]

  const int tid   = threadIdx.x;
  const int lane  = tid & 31;
  const int wave  = tid >> 5;          // 0..3
  const int wr    = (wave >> 1) * 32;  // wave row offset in tile
  const int wc    = (wave & 1)  * 32;  // wave col offset in tile
  const int l16   = lane & 15;
  const int khalf = lane >> 4;
  const int mBase = blockIdx.y * TM;
  const int nBase = blockIdx.x * TN;

  // A staging: thread -> (row group, float4 K-offset); 4 row-iterations
  const int rowSA = tid >> 3;          // 0..15  (+16*rr)
  const int colSA = (tid & 7) * 4;     // 0..28
  // B staging: thread -> (row group, bf16x8 K-offset); 2 row-iterations
  const int rowSB = tid >> 2;          // 0..31  (+32*rr)
  const int colSB = (tid & 3) * 8;     // 0..24

  float4 av[4];
  bf16x8 bvh[2];

  auto loadA = [&](int k0) {
    const bool kfull = (k0 + TK) <= Kd;
#pragma unroll
    for (int rr = 0; rr < 4; ++rr) {
      int gm = mBase + rowSA + 16 * rr;
      int gk = k0 + colSA;
      float4 v = make_float4(0.f, 0.f, 0.f, 0.f);
      if (gm < M) {
        const float* p = A + (size_t)gm * Kd + gk;
        if (kfull) {
          v = *(const float4*)p;               // unguarded b128 fast path
        } else {
          if (gk + 0 < Kd) v.x = p[0];
          if (gk + 1 < Kd) v.y = p[1];
          if (gk + 2 < Kd) v.z = p[2];
          if (gk + 3 < Kd) v.w = p[3];
        }
      }
      av[rr] = v;
    }
  };

  auto loadB = [&](int k0) {
    const bool kfull = (k0 + TK) <= Kd;
#pragma unroll
    for (int rr = 0; rr < 2; ++rr) {
      int gn = nBase + rowSB + 32 * rr;
      int gk = k0 + colSB;
      bf16x8 v;
#pragma unroll
      for (int i = 0; i < 8; ++i) v[i] = (__bf16)0.0f;
      if (gn < Nc) {
        const __bf16* p = B + (size_t)gn * Kd + gk;
        if (kfull) {
          v = *(const bf16x8*)p;               // 16B aligned (K % 8 == 0)
        } else {
#pragma unroll
          for (int i = 0; i < 8; ++i)
            if (gk + i < Kd) v[i] = p[i];
        }
      }
      bvh[rr] = v;
    }
  };

  auto storeAB = [&](int buf) {
#pragma unroll
    for (int rr = 0; rr < 4; ++rr) {
      bf16x4 h;
      h[0] = (__bf16)av[rr].x; h[1] = (__bf16)av[rr].y;
      h[2] = (__bf16)av[rr].z; h[3] = (__bf16)av[rr].w;
      *(bf16x4*)&Al[buf][rowSA + 16 * rr][colSA] = h;   // ds_store_b64
    }
#pragma unroll
    for (int rr = 0; rr < 2; ++rr)
      *(bf16x8*)&Bl[buf][rowSB + 32 * rr][colSB] = bvh[rr];  // ds_store_b128
  };

  f32x8 acc[2][2];
#pragma unroll
  for (int a = 0; a < 2; ++a)
#pragma unroll
    for (int b = 0; b < 2; ++b)
#pragma unroll
      for (int q = 0; q < 8; ++q) acc[a][b][q] = 0.0f;

  // prologue: stage first tile
  loadA(0); loadB(0); storeAB(0);
  int cur = 0;

  for (int k0 = 0; k0 < Kd; k0 += TK) {
    __syncthreads();
    const bool hasNext = (k0 + TK) < Kd;
    if (hasNext) { loadA(k0 + TK); loadB(k0 + TK); }   // loads in flight

    bf16x16 af[2], bfg[2];
#pragma unroll
    for (int mi = 0; mi < 2; ++mi) {
      int m = wr + mi * 16 + l16;
#pragma unroll
      for (int j = 0; j < 8; ++j) {          // ISA 16-bit A 16x32 layout
        af[mi][j]     = Al[cur][m][khalf * 8 + j];
        af[mi][8 + j] = Al[cur][m][16 + khalf * 8 + j];
      }
    }
#pragma unroll
    for (int ni = 0; ni < 2; ++ni) {
      int n = wc + ni * 16 + l16;
#pragma unroll
      for (int e = 0; e < 8; ++e) {           // ISA 16-bit B 32x16 layout
        bfg[ni][e]     = Bl[cur][n][khalf * 8 + e];
        bfg[ni][8 + e] = Bl[cur][n][16 + khalf * 8 + e];
      }
    }
#pragma unroll
    for (int mi = 0; mi < 2; ++mi)
#pragma unroll
      for (int ni = 0; ni < 2; ++ni)
        acc[mi][ni] = __builtin_amdgcn_wmma_f32_16x16x32_bf16(
            false, af[mi], false, bfg[ni], (short)0, acc[mi][ni], false, false);

    if (hasNext) { storeAB(cur ^ 1); cur ^= 1; }
  }

  // epilogue (C/D f32 layout: row = 8*khalf + r, col = lane&15)
#pragma unroll
  for (int mi = 0; mi < 2; ++mi) {
#pragma unroll
    for (int ni = 0; ni < 2; ++ni) {
      int gc = nBase + wc + ni * 16 + l16;
#pragma unroll
      for (int r = 0; r < 8; ++r) {
        int gr = mBase + wr + mi * 16 + khalf * 8 + r;
        if (gr < M && gc < Nc) {
          float v = acc[mi][ni][r];
          float res;
          if constexpr (MODE == 0) {
            res = v + bias[gc];
          } else if constexpr (MODE == 1) {
            float t = v + bias[gc];
            t = t * (gamma[gc] * rsqrtf(1.0f + BNE)) + beta[gc];
            res = (t > 0.0f) ? t : expm1f(t);
          } else {
            float dx = sp[2 * gr]     - sp[2 * gc];
            float dy = sp[2 * gr + 1] - sp[2 * gc + 1];
            res = 2.0f * v - rowN[gr] - colN[gc] - (dx * dx + dy * dy);
          }
          C[(size_t)gr * Nc + gc] = res;
          if constexpr (MODE == 1) {
            if (C2) C2[(size_t)gr * Nc + gc] = res;
          }
        }
      }
    }
  }
}

// ---------------- conversion kernels ----------------
// f32 [K,Nc] -> bf16 [Nc,K] (transpose); coalesced read
__global__ void k_conv_bt(const float* __restrict__ in, __bf16* __restrict__ out,
                          int Kd, int Nc) {
  long t = (long)blockIdx.x * blockDim.x + threadIdx.x;
  if (t >= (long)Kd * Nc) return;
  int k = (int)(t / Nc), n = (int)(t % Nc);
  out[(size_t)n * Kd + k] = (__bf16)in[t];
}
// f32 -> bf16, same layout
__global__ void k_conv_bf16(const float* __restrict__ in, __bf16* __restrict__ out, long n) {
  long i = (long)blockIdx.x * blockDim.x + threadIdx.x;
  if (i < n) out[i] = (__bf16)in[i];
}

// ---------------- small kernels ----------------
__global__ void k_fill_f(float* p, float v, long n) {
  long i = (long)blockIdx.x * blockDim.x + threadIdx.x;
  if (i < n) p[i] = v;
}
__global__ void k_fill_u(unsigned* p, unsigned v, long n) {
  long i = (long)blockIdx.x * blockDim.x + threadIdx.x;
  if (i < n) p[i] = v;
}
__global__ void k_copy_block(float* dst, int dstride, const float* src, int sstride,
                             int rows, int cols) {
  long t = (long)blockIdx.x * blockDim.x + threadIdx.x;
  long total = (long)rows * cols;
  if (t >= total) return;
  int r = (int)(t / cols), c = (int)(t % cols);
  dst[(size_t)r * dstride + c] = src[(size_t)r * sstride + c];
}
__global__ void k_relu(float* p, long n) {
  long i = (long)blockIdx.x * blockDim.x + threadIdx.x;
  if (i < n) p[i] = fmaxf(p[i], 0.0f);
}
__global__ void k_row_norm(const float* z, float* out, int rows, int cols) {
  int i = blockIdx.x * blockDim.x + threadIdx.x;
  if (i >= rows) return;
  float s = 0.0f;
  for (int d = 0; d < cols; ++d) { float t = z[(size_t)i * cols + d]; s += t * t; }
  out[i] = s;
}

// ---- edge attention ----
__global__ void k_edge_logits(const float* __restrict__ q, const float* __restrict__ k,
                              const int* __restrict__ src, const int* __restrict__ dst,
                              float* __restrict__ logit, unsigned* __restrict__ mmax,
                              float sc) {
  int e = blockIdx.x * blockDim.x + threadIdx.x;
  if (e >= Ee) return;
  const float4* qr = (const float4*)(q + (size_t)dst[e] * GHc);
  const float4* kr = (const float4*)(k + (size_t)src[e] * GHc);
  float acc = 0.0f;
#pragma unroll 8
  for (int i = 0; i < GHc / 4; ++i) {
    float4 a = qr[i], b = kr[i];
    acc += a.x * b.x + a.y * b.y + a.z * b.z + a.w * b.w;
  }
  acc *= sc;
  logit[e] = acc;
  atomicMax(mmax + dst[e], f2ord(acc));
}
__global__ void k_edge_exp(float* __restrict__ logit, const unsigned* __restrict__ mmax,
                           float* __restrict__ ssum, const int* __restrict__ dst) {
  int e = blockIdx.x * blockDim.x + threadIdx.x;
  if (e >= Ee) return;
  float m = ord2f(mmax[dst[e]]);
  float ex = expf(logit[e] - m);
  logit[e] = ex;
  atomicAdd(ssum + dst[e], ex);
}
__global__ void k_edge_div(float* __restrict__ alpha, const float* __restrict__ ssum,
                           const int* __restrict__ dst) {
  int e = blockIdx.x * blockDim.x + threadIdx.x;
  if (e >= Ee) return;
  alpha[e] = alpha[e] / ssum[dst[e]];
}
__global__ void k_scatter_attn(float* __restrict__ out, const int* __restrict__ src,
                               const int* __restrict__ dst, const float* __restrict__ alpha,
                               const float* __restrict__ v) {
  long t = (long)blockIdx.x * blockDim.x + threadIdx.x;
  if (t >= (long)Ee * GHc) return;
  int e = (int)(t >> 7), d = (int)(t & (GHc - 1));
  atomicAdd(&out[(size_t)dst[e] * GHc + d], alpha[e] * v[(size_t)src[e] * GHc + d]);
}
// (A @ o) as SpMM scatter: row src[e] += w * alpha[e] * o[dst[e]]  (src!=dst masked)
__global__ void k_scatter_h(float* __restrict__ hnew, const int* __restrict__ src,
                            const int* __restrict__ dst, const float* __restrict__ alpha,
                            const float* __restrict__ o, float w) {
  long t = (long)blockIdx.x * blockDim.x + threadIdx.x;
  if (t >= (long)Ee * GHc) return;
  int e = (int)(t >> 7), d = (int)(t & (GHc - 1));
  int s = src[e], dd = dst[e];
  if (s == dd) return;
  atomicAdd(&hnew[(size_t)s * GHc + d], w * alpha[e] * o[(size_t)dd * GHc + d]);
}
__global__ void k_dense_ap(float* __restrict__ at, const int* __restrict__ src,
                           const int* __restrict__ dst, const float* __restrict__ alpha) {
  int e = blockIdx.x * blockDim.x + threadIdx.x;
  if (e >= Ee) return;
  int s = src[e], dd = dst[e];
  if (s != dd) atomicAdd(&at[(size_t)s * Nn + dd], alpha[e]);
}

// ---- top-K per row (wave per row, per-lane top-10 + LDS merge) ----
__global__ void k_topk_rows(const float* __restrict__ G, float* __restrict__ tv,
                            int* __restrict__ ti) {
  int row = blockIdx.x, lane = threadIdx.x;
  const float* g = G + (size_t)row * Nn;
  float v[KT]; int ix[KT];
#pragma unroll
  for (int t = 0; t < KT; ++t) { v[t] = -INFINITY; ix[t] = 0x7FFFFFFF; }
  for (int j = lane; j < Nn; j += 32) {
    float x = g[j];
    if (x > v[KT - 1]) {
      int p = KT - 1;
      while (p > 0 && v[p - 1] < x) { v[p] = v[p - 1]; ix[p] = ix[p - 1]; --p; }
      v[p] = x; ix[p] = j;
    }
  }
  __shared__ float sv[32 * KT];
  __shared__ int   si[32 * KT];
  for (int t = 0; t < KT; ++t) { sv[lane * KT + t] = v[t]; si[lane * KT + t] = ix[t]; }
  __syncthreads();
  if (lane == 0) {
    for (int t = 0; t < KT; ++t) {
      float best = -INFINITY; int bi = 0x7FFFFFFF, bp = 0;
      for (int p = 0; p < 32 * KT; ++p) {
        float x = sv[p]; int id = si[p];
        if (x > best || (x == best && id < bi)) { best = x; bi = id; bp = p; }
      }
      tv[(size_t)row * KT + t] = best;
      ti[(size_t)row * KT + t] = bi;
      sv[bp] = -INFINITY; si[bp] = 0x7FFFFFFF;
    }
  }
}

// ---- z_out / x1_hr (block per row; 256 threads over latent dim) ----
__global__ void k_finalize_rows(const float* __restrict__ z, const float* __restrict__ znb,
                                const float* __restrict__ tv, const int* __restrict__ ti,
                                float* __restrict__ zout, float* __restrict__ x1hr) {
  int row = blockIdx.x, d = threadIdx.x;
  float vals[KT]; int idx[KT];
#pragma unroll
  for (int t = 0; t < KT; ++t) { vals[t] = tv[(size_t)row * KT + t]; idx[t] = ti[(size_t)row * KT + t]; }
  float m = vals[0];
  float e[KT];
#pragma unroll
  for (int t = 0; t < KT; ++t) e[t] = expf(vals[t] - m);
  float den9 = 0.0f;
#pragma unroll
  for (int t = 0; t < SCLc; ++t) den9 += e[t];
  float acc = z[(size_t)row * LATc + d];
#pragma unroll
  for (int t = 0; t < SCLc; ++t) {
    float w  = e[t] / den9;
    float xh = znb[(size_t)idx[t] * LATc + d];
    acc += w * xh;
    x1hr[((size_t)row * SCLc + t) * LATc + d] = xh * w;
  }
  zout[(size_t)row * LATc + d] = acc;
}

// ---- soft cluster assignment q ----
__global__ void k_cluster_q(const float* __restrict__ zout, const float* __restrict__ cl,
                            float* __restrict__ qout) {
  int i = blockIdx.x, c = threadIdx.x;
  __shared__ float qs[NCc];
  if (c < NCc) {
    float d2 = 0.0f;
    for (int d = 0; d < LATc; ++d) {
      float t = zout[(size_t)i * LATc + d] - cl[(size_t)c * LATc + d];
      d2 += t * t;
    }
    qs[c] = 1.0f / (1.0f + d2);   // ALPHA=1 -> exponent 1
  }
  __syncthreads();
  if (c == 0) {
    float s = 0.0f;
    for (int t = 0; t < NCc; ++t) s += qs[t];
    for (int t = 0; t < NCc; ++t) qout[(size_t)i * NCc + t] = qs[t] / s;
  }
}

// ---------------- host-side orchestration ----------------
static void launch_gemm_bias(const float* A, const __bf16* Bt, const float* bias, float* C,
                             int M, int Kd, int Nc, hipStream_t st) {
  dim3 g(cdiv(Nc, TN), cdiv(M, TM));
  gemm_wmma<0><<<g, 128, 0, st>>>(A, Bt, bias, nullptr, nullptr, C, nullptr,
                                  nullptr, nullptr, nullptr, M, Kd, Nc);
}
static void launch_gemm_bnelu(const float* A, const __bf16* Bt, const float* bias,
                              const float* ga, const float* be, float* C, float* C2,
                              int M, int Kd, int Nc, hipStream_t st) {
  dim3 g(cdiv(Nc, TN), cdiv(M, TM));
  gemm_wmma<1><<<g, 128, 0, st>>>(A, Bt, bias, ga, be, C, C2,
                                  nullptr, nullptr, nullptr, M, Kd, Nc);
}
static void launch_gemm_score(const float* A, const __bf16* Bt, const float* rowN,
                              const float* colN, const float* sp, float* C,
                              int M, int Kd, int Nc, hipStream_t st) {
  dim3 g(cdiv(Nc, TN), cdiv(M, TM));
  gemm_wmma<2><<<g, 128, 0, st>>>(A, Bt, nullptr, nullptr, nullptr, C, nullptr,
                                  rowN, colN, sp, M, Kd, Nc);
}

static void run_tconv(const float* hin,
                      const __bf16* Wqt, const __bf16* Wkt, const __bf16* Wvt,
                      const __bf16* Wst,
                      const float* bq, const float* bk, const float* bv, const float* bs,
                      const int* src, const int* dst,
                      float* qb, float* kb, float* vb, float* outb,
                      float* alpha, float* sbuf, unsigned* mbits, hipStream_t st) {
  launch_gemm_bias(hin, Wqt, bq, qb,   Nn, GHc, GHc, st);
  launch_gemm_bias(hin, Wkt, bk, kb,   Nn, GHc, GHc, st);
  launch_gemm_bias(hin, Wvt, bv, vb,   Nn, GHc, GHc, st);
  launch_gemm_bias(hin, Wst, bs, outb, Nn, GHc, GHc, st);  // out starts as x@Ws + bs
  k_fill_u<<<cdiv(Nn, 256), 256, 0, st>>>(mbits, 0x007FFFFFu /* ord(-inf) */, Nn);
  k_fill_f<<<cdiv(Nn, 256), 256, 0, st>>>(sbuf, 0.0f, Nn);
  const float sc = 1.0f / sqrtf((float)GHc);
  k_edge_logits<<<cdiv(Ee, 256), 256, 0, st>>>(qb, kb, src, dst, alpha, mbits, sc);
  k_edge_exp<<<cdiv(Ee, 256), 256, 0, st>>>(alpha, mbits, sbuf, dst);
  k_edge_div<<<cdiv(Ee, 256), 256, 0, st>>>(alpha, sbuf, dst);
  k_scatter_attn<<<cdiv((long)Ee * GHc, 256), 256, 0, st>>>(outb, src, dst, alpha, vb);
}

static void run_encode(const float* x,
                       const __bf16* W1t, const float* b1, const float* g1, const float* be1,
                       const __bf16* W2t, const float* b2, const float* g2, const float* be2,
                       const __bf16* cWt /* 12 x GH*GH: l-major, q,k,v,s */,
                       const float* cbq, const float* cbk, const float* cbv, const float* cbs,
                       const int* asrc, const int* adst, const int* psrc, const int* pdst,
                       float* f1, float* ft, float* hA, float* hB,
                       float* qb, float* kb, float* vb, float* ob, float* opb,
                       float* alpha, float* sbuf, unsigned* mbits,
                       float* zdst /* N x LAT */, float* atten /* or null */,
                       hipStream_t st) {
  // feature encoder
  launch_gemm_bnelu(x,  W1t, b1, g1, be1, f1, nullptr, Nn, DINc, FH1c, st);
  launch_gemm_bnelu(f1, W2t, b2, g2, be2, ft, nullptr, Nn, FH1c, FH2c, st);
  // z[:, 0:128] = feat
  k_copy_block<<<cdiv((long)Nn * FH2c, 256), 256, 0, st>>>(zdst, LATc, ft, FH2c, Nn, FH2c);

  const float* h = ft;
  float* bufs[3] = {hA, hB, hA};
  for (int l = 0; l < 3; ++l) {
    float* hnew = bufs[l];
    k_fill_f<<<cdiv((long)Nn * GHc, 256), 256, 0, st>>>(hnew, 0.0f, (long)Nn * GHc);
    const __bf16* Wqt = cWt + (size_t)(4 * l + 0) * GHc * GHc;
    const __bf16* Wkt = cWt + (size_t)(4 * l + 1) * GHc * GHc;
    const __bf16* Wvt = cWt + (size_t)(4 * l + 2) * GHc * GHc;
    const __bf16* Wst = cWt + (size_t)(4 * l + 3) * GHc * GHc;
    const float* bq = cbq + (size_t)l * GHc;
    const float* bk = cbk + (size_t)l * GHc;
    const float* bv = cbv + (size_t)l * GHc;
    const float* bs = cbs + (size_t)l * GHc;

    // tconv on adj -> ob, alpha; h_new += (1-AT) * (A @ o)
    run_tconv(h, Wqt, Wkt, Wvt, Wst, bq, bk, bv, bs, asrc, adst,
              qb, kb, vb, ob, alpha, sbuf, mbits, st);
    k_scatter_h<<<cdiv((long)Ee * GHc, 256), 256, 0, st>>>(hnew, asrc, adst, alpha, ob,
                                                           1.0f - ATc);
    // tconv on adj_prue -> opb, alpha; h_new += AT * (Ap @ op)
    const float* hp = (l == 1) ? ob : h;
    run_tconv(hp, Wqt, Wkt, Wvt, Wst, bq, bk, bv, bs, psrc, pdst,
              qb, kb, vb, opb, alpha, sbuf, mbits, st);
    k_scatter_h<<<cdiv((long)Ee * GHc, 256), 256, 0, st>>>(hnew, psrc, pdst, alpha, opb, ATc);

    if (l == 2 && atten) {   // atten = dense lastAp (first encode only)
      k_fill_f<<<cdiv((long)Nn * Nn, 256), 256, 0, st>>>(atten, 0.0f, (long)Nn * Nn);
      k_dense_ap<<<cdiv(Ee, 256), 256, 0, st>>>(atten, psrc, pdst, alpha);
    }
    if (l < 2)
      k_relu<<<cdiv((long)Nn * GHc, 256), 256, 0, st>>>(hnew, (long)Nn * GHc);
    h = hnew;
  }
  // z[:, 128:256] = mu
  k_copy_block<<<cdiv((long)Nn * GHc, 256), 256, 0, st>>>(zdst + FH2c, LATc, h, GHc, Nn, GHc);
}

extern "C" void kernel_launch(void* const* d_in, const int* in_sizes, int n_in,
                              void* d_out, int out_size, void* d_ws, size_t ws_size,
                              hipStream_t stream) {
  // ---- inputs (setup_inputs dict order) ----
  const float* x    = (const float*)d_in[0];
  const float* xnb  = (const float*)d_in[1];
  const float* sp   = (const float*)d_in[2];
  const int*   adj  = (const int*)d_in[3];    // (2,E): src=adj, dst=adj+E
  const int*   adjp = (const int*)d_in[4];
  const float* eW1  = (const float*)d_in[5];
  const float* eb1  = (const float*)d_in[6];
  const float* eg1  = (const float*)d_in[7];
  const float* ebe1 = (const float*)d_in[8];
  const float* eW2  = (const float*)d_in[9];
  const float* eb2  = (const float*)d_in[10];
  const float* eg2  = (const float*)d_in[11];
  const float* ebe2 = (const float*)d_in[12];
  const float* cWq  = (const float*)d_in[13];
  const float* cWk  = (const float*)d_in[14];
  const float* cWv  = (const float*)d_in[15];
  const float* cWs  = (const float*)d_in[16];
  const float* cbq  = (const float*)d_in[17];
  const float* cbk  = (const float*)d_in[18];
  const float* cbv  = (const float*)d_in[19];
  const float* cbs  = (const float*)d_in[20];
  const float* dW   = (const float*)d_in[21];
  const float* db   = (const float*)d_in[22];
  const float* dg   = (const float*)d_in[23];
  const float* dbe  = (const float*)d_in[24];
  const float* clus = (const float*)d_in[25];

  const int* asrc = adj,  *adst = adj  + Ee;
  const int* psrc = adjp, *pdst = adjp + Ee;

  // ---- outputs (flat concat, return order) ----
  float* out    = (float*)d_out;
  float* o_zout = out;
  float* o_def1 = o_zout + (size_t)Nn * LATc;
  float* o_q    = o_def1 + (size_t)Nn * DINc;
  float* o_ftx  = o_q    + (size_t)Nn * NCc;
  float* o_def2 = o_ftx  + (size_t)Nn * FH2c;
  float* o_x1   = o_def2 + (size_t)Nn * DINc;
  float* o_att  = o_x1   + (size_t)Nn * SCLc * LATc;

  // ---- workspace (f32 region) ----
  float* ws = (float*)d_ws;
  size_t o = 0;
  float* G    = ws + o; o += (size_t)Nn * Nn;        // -(df+ds) score matrix
  float* zb   = ws + o; o += (size_t)Nn * LATc;      // z  = [feat_x | mu]
  float* znb  = ws + o; o += (size_t)Nn * LATc;      // z_nb
  float* f1   = ws + o; o += (size_t)Nn * FH1c;
  float* ft   = ws + o; o += (size_t)Nn * FH2c;
  float* hA   = ws + o; o += (size_t)Nn * GHc;
  float* hB   = ws + o; o += (size_t)Nn * GHc;
  float* qb   = ws + o; o += (size_t)Nn * GHc;
  float* kb   = ws + o; o += (size_t)Nn * GHc;
  float* vb   = ws + o; o += (size_t)Nn * GHc;
  float* ob   = ws + o; o += (size_t)Nn * GHc;
  float* opb  = ws + o; o += (size_t)Nn * GHc;
  float* alpha= ws + o; o += (size_t)Ee;
  float* sbuf = ws + o; o += (size_t)Nn;
  unsigned* mbits = (unsigned*)(ws + o); o += (size_t)Nn;
  float* zz   = ws + o; o += (size_t)Nn;
  float* z2   = ws + o; o += (size_t)Nn;
  float* tv   = ws + o; o += (size_t)Nn * KT;
  int*   ti   = (int*)(ws + o); o += (size_t)Nn * KT;

  // ---- workspace (bf16 region; 16B aligned since o is a multiple of 4) ----
  __bf16* bfb = (__bf16*)(ws + o);
  size_t ob16 = 0;
  __bf16* W1t  = bfb + ob16; ob16 += (size_t)FH1c * DINc;   // [256,3000]
  __bf16* W2t  = bfb + ob16; ob16 += (size_t)FH2c * FH1c;   // [128,256]
  __bf16* cWt  = bfb + ob16; ob16 += (size_t)12 * GHc * GHc;// 12 x [128,128]
  __bf16* dWt  = bfb + ob16; ob16 += (size_t)DINc * LATc;   // [3000,256]
  __bf16* znbB = bfb + ob16; ob16 += (size_t)Nn * LATc;     // [6000,256]

  // ---- pre-convert all GEMM B operands to bf16 [N,K] ----
  k_conv_bt<<<cdiv((long)DINc * FH1c, 256), 256, 0, stream>>>(eW1, W1t, DINc, FH1c);
  k_conv_bt<<<cdiv((long)FH1c * FH2c, 256), 256, 0, stream>>>(eW2, W2t, FH1c, FH2c);
  for (int l = 0; l < 3; ++l) {
    const float* Wm[4] = {cWq + (size_t)l * GHc * GHc, cWk + (size_t)l * GHc * GHc,
                          cWv + (size_t)l * GHc * GHc, cWs + (size_t)l * GHc * GHc};
    for (int m = 0; m < 4; ++m)
      k_conv_bt<<<cdiv((long)GHc * GHc, 256), 256, 0, stream>>>(
          Wm[m], cWt + (size_t)(4 * l + m) * GHc * GHc, GHc, GHc);
  }
  k_conv_bt<<<cdiv((long)LATc * DINc, 256), 256, 0, stream>>>(dW, dWt, LATc, DINc);

  // ---- encode x (writes z and dense atten) ----
  run_encode(x, W1t, eb1, eg1, ebe1, W2t, eb2, eg2, ebe2,
             cWt, cbq, cbk, cbv, cbs,
             asrc, adst, psrc, pdst,
             f1, ft, hA, hB, qb, kb, vb, ob, opb, alpha, sbuf, mbits,
             zb, o_att, stream);
  // feat_x output = z[:, 0:128]
  k_copy_block<<<cdiv((long)Nn * FH2c, 256), 256, 0, stream>>>(o_ftx, FH2c, zb, LATc, Nn, FH2c);

  // ---- encode x_neighbor (writes z_nb) ----
  run_encode(xnb, W1t, eb1, eg1, ebe1, W2t, eb2, eg2, ebe2,
             cWt, cbq, cbk, cbv, cbs,
             asrc, adst, psrc, pdst,
             f1, ft, hA, hB, qb, kb, vb, ob, opb, alpha, sbuf, mbits,
             znb, nullptr, stream);

  // ---- pairwise score G = -(df + ds) via WMMA NT-GEMM + fused epilogue ----
  k_conv_bf16<<<cdiv((long)Nn * LATc, 256), 256, 0, stream>>>(znb, znbB, (long)Nn * LATc);
  k_row_norm<<<cdiv(Nn, 256), 256, 0, stream>>>(zb,  zz, Nn, LATc);
  k_row_norm<<<cdiv(Nn, 256), 256, 0, stream>>>(znb, z2, Nn, LATc);
  launch_gemm_score(zb, znbB, zz, z2, sp, G, Nn, LATc, Nn, stream);

  // ---- top-K, adaptive fusion ----
  k_topk_rows<<<Nn, 32, 0, stream>>>(G, tv, ti);
  k_finalize_rows<<<Nn, LATc, 0, stream>>>(zb, znb, tv, ti, o_zout, o_x1);

  // ---- decoder (writes de_feat twice via fused epilogue) ----
  launch_gemm_bnelu(o_zout, dWt, db, dg, dbe, o_def1, o_def2, Nn, LATc, DINc, stream);

  // ---- cluster soft-assignment ----
  k_cluster_q<<<Nn, 32, 0, stream>>>(o_zout, clus, o_q);
}
